// ptBEVnet_84430467104794
// MI455X (gfx1250) — compile-verified, matrix operand order checked by reading, not run on hardware
//
#include <hip/hip_runtime.h>

#define BB 2
#define HH 480
#define WWID 360
#define NSEG (BB*HH*WWID)     // 345600
#define MAXPT 256
#define EPSBN 1e-5f
#define SLOPE 0.01f

typedef __attribute__((ext_vector_type(16))) _Float16 v16h;
typedef __attribute__((ext_vector_type(8)))  float    v8f;

// ---- monotone encoding: float -> uint so that uint-max == float-max ----
__device__ __forceinline__ unsigned f2mono(float f) {
  unsigned u = __float_as_uint(f);
  return (u & 0x80000000u) ? ~u : (u | 0x80000000u);
}
__device__ __forceinline__ float mono2f(unsigned e) {
  unsigned u = (e & 0x80000000u) ? (e ^ 0x80000000u) : ~e;
  return __uint_as_float(u);
}

// 16-bit A/B fragment K-index map for WMMA 16x16x32 (CDNA5 ISA 7.12.2):
// lanes 0-15 hold K 0..7 (V0..V3) and 16..23 (V4..V7); lanes 16-31 hold +8.
__device__ __forceinline__ int kmap(int e, int half_) {
  int v = e >> 1, o = e & 1;
  return (v < 4) ? (2*v + o + 8*half_) : (16 + 2*(v - 4) + o + 8*half_);
}

// ------------------------------------------------------------------ init ----
__global__ void init_ws(unsigned* __restrict__ p, long long words) {
  long long i = (long long)blockIdx.x * blockDim.x + threadIdx.x;
  long long stride = (long long)gridDim.x * blockDim.x;
  for (; i < words; i += stride) p[i] = 0u;
}

// ------------------------------------------------- weight prep (BN folds) ----
// Writes f16 weights pre-swizzled into B-fragment order:
//   wdst[(((kc*NT + nt)*32 + lane)*16 + e)] = W[kc*32 + kmap(e,lane>>4)][nt*16 + lane&15]
// with optional per-input (bn-pre) and per-output (bn-post) scale folds.
__global__ void prep_layer(const float* __restrict__ Wsrc, const float* __restrict__ bsrc,
                           int Kreal, int KC, int NT, int Nout,
                           const float* bnPreG, const float* bnPreB,
                           const float* bnPreM, const float* bnPreV,
                           const float* bnPostG, const float* bnPostB,
                           const float* bnPostM, const float* bnPostV,
                           _Float16* __restrict__ wdst, float* __restrict__ bdst) {
  int tid = blockIdx.x * blockDim.x + threadIdx.x;
  int fragCount = KC * NT * 512;
  if (tid < fragCount) {
    int e = tid & 15, lane = (tid >> 4) & 31, rest = tid >> 9;
    int nt = rest % NT, kc = rest / NT;
    int half_ = lane >> 4, n = nt * 16 + (lane & 15);
    int k = kc * 32 + kmap(e, half_);
    float val = 0.f;
    if (k < Kreal && n < Nout) {
      val = Wsrc[k * Nout + n];
      if (bnPreG)  val *= bnPreG[k] * rsqrtf(bnPreV[k] + EPSBN);
      if (bnPostG) val *= bnPostG[n] * rsqrtf(bnPostV[n] + EPSBN);
    }
    wdst[tid] = (_Float16)val;
  }
  if (tid < Nout) {
    float bb = bsrc[tid];
    if (bnPreG) {  // layer-1 only: shift term t0 @ W
      for (int k = 0; k < Kreal; ++k) {
        float s = bnPreG[k] * rsqrtf(bnPreV[k] + EPSBN);
        float t = bnPreB[k] - bnPreM[k] * s;
        bb += t * Wsrc[k * Nout + tid];
      }
    }
    if (bnPostG) {
      float s = bnPostG[tid] * rsqrtf(bnPostV[tid] + EPSBN);
      float t = bnPostB[tid] - bnPostM[tid] * s;
      bb = bb * s + t;
    }
    bdst[tid] = bb;
  }
}

// ------------------------------------------------------------ MLP helpers ----
__device__ __forceinline__ v16h load_a(const _Float16* act, int stride, int lane, int kbase) {
  int r = lane & 15, half_ = lane >> 4;
  v16h a;
#pragma unroll
  for (int e = 0; e < 16; ++e) a[e] = act[r * stride + kbase + kmap(e, half_)];
  return a;
}

template <int KC, int NT>
__device__ __forceinline__ void layerN(const _Float16* __restrict__ in, int is,
                                       _Float16* __restrict__ out, int os,
                                       const _Float16* __restrict__ wf,
                                       const float* __restrict__ bias, int lane) {
  int col = lane & 15, mbase = (lane >> 4) * 8;
  for (int nt = 0; nt < NT; ++nt) {
    v8f c = {};
#pragma unroll
    for (int kc = 0; kc < KC; ++kc) {
      v16h a  = load_a(in, is, lane, kc * 32);
      v16h bw = *(const v16h*)(wf + ((size_t)((kc * NT + nt) * 32 + lane) << 4));
      c = __builtin_amdgcn_wmma_f32_16x16x32_f16(false, a, false, bw, (short)0, c, false, false);
    }
    int n = nt * 16 + col;
    float bb = bias[n];
#pragma unroll
    for (int v = 0; v < 8; ++v) {
      float y = c[v] + bb;
      y = (y >= 0.f) ? y : SLOPE * y;              // LeakyReLU (BN already folded)
      out[(mbase + v) * os + n] = (_Float16)y;
    }
  }
}

// ------------------------------------- fused MLP + atomic scatter-max -------
#define MLP_WAVES 4
__global__ void __launch_bounds__(128) mlp_scatter(
    const float* __restrict__ pt_fea, const int* __restrict__ xy,
    const int* __restrict__ bidx, int N,
    unsigned* __restrict__ cnt, unsigned* __restrict__ maxenc,
    const _Float16* __restrict__ wf1, const _Float16* __restrict__ wf2,
    const _Float16* __restrict__ wf3, const _Float16* __restrict__ wf4,
    const float* __restrict__ bz1, const float* __restrict__ bz2,
    const float* __restrict__ bz3, const float* __restrict__ bz4) {
  __shared__ _Float16 sA[MLP_WAVES][16 * 128];
  __shared__ _Float16 sB[MLP_WAVES][16 * 256];
  __shared__ unsigned sKey[MLP_WAVES][16];
  int wid = threadIdx.x >> 5, lane = threadIdx.x & 31;
  int tile = blockIdx.x * MLP_WAVES + wid;

  // voxel key + within-voxel rank (cap at MAXPT); sentinel key for dropped
  if (lane < 16) {
    unsigned key = 0xFFFFFFFFu;
    int p = tile * 16 + lane;
    if (p < N) {
      int b = bidx[p], xi = xy[2 * p], yi = xy[2 * p + 1];
      unsigned k2 = (unsigned)((b * HH + xi) * WWID + yi);
      unsigned r = atomicAdd(&cnt[k2], 1u);
      if (r < MAXPT) key = k2;
    }
    sKey[wid][lane] = key;
  }
  // zero-pad layer-1 staging (K padded 9 -> 32; avoids NaN*0 in WMMA)
  for (int i = lane; i < 16 * 32; i += 32)
    sA[wid][(i >> 5) * 128 + (i & 31)] = (_Float16)0.f;
  __syncthreads();
  for (int i = lane; i < 16 * 9; i += 32) {
    int r = i / 9, k = i - r * 9, p = tile * 16 + r;
    if (p < N) sA[wid][r * 128 + k] = (_Float16)pt_fea[p * 9 + k];
  }
  __syncthreads();

  layerN<1, 4>(sA[wid], 128, sB[wid], 256, wf1, bz1, lane);   // 9->64
  __syncthreads();
  layerN<2, 8>(sB[wid], 256, sA[wid], 128, wf2, bz2, lane);   // 64->128
  __syncthreads();
  layerN<4, 16>(sA[wid], 128, sB[wid], 256, wf3, bz3, lane);  // 128->256
  __syncthreads();

  // layer 4 (256->512, no activation) fused with scatter-max
  int col = lane & 15, mbase = (lane >> 4) * 8;
  for (int nt = 0; nt < 32; ++nt) {
    v8f c = {};
#pragma unroll
    for (int kc = 0; kc < 8; ++kc) {
      v16h a  = load_a(sB[wid], 256, lane, kc * 32);
      v16h bw = *(const v16h*)(wf4 + ((size_t)((kc * 32 + nt) * 32 + lane) << 4));
      c = __builtin_amdgcn_wmma_f32_16x16x32_f16(false, a, false, bw, (short)0, c, false, false);
    }
    int ch = nt * 16 + col;
    float bb = bz4[ch];
#pragma unroll
    for (int v = 0; v < 8; ++v) {
      unsigned key = sKey[wid][mbase + v];
      if (key != 0xFFFFFFFFu)
        atomicMax(&maxenc[(size_t)key * 512 + ch], f2mono(c[v] + bb));
    }
  }
}

// -------------------------------- per-voxel 512->32 WMMA GEMM + LReLU -------
#define BEV_WAVES 4
__global__ void __launch_bounds__(128) bev_gemm(
    const unsigned* __restrict__ cnt, const unsigned* __restrict__ maxenc,
    const _Float16* __restrict__ wfo, const float* __restrict__ bzo,
    float* __restrict__ out) {
  __shared__ unsigned occL[BEV_WAVES][16];
  int wid = threadIdx.x >> 5, lane = threadIdx.x & 31;
  int tile = blockIdx.x * BEV_WAVES + wid;  // NSEG divisible by 64: no tail
  int v0 = tile * 16;

  if (lane < 16) occL[wid][lane] = cnt[v0 + lane];
  __syncthreads();

  int col = lane & 15, mbase = (lane >> 4) * 8;
  int r = lane & 15, half_ = lane >> 4;
  size_t rowBase = (size_t)(v0 + r) * 512;
  bool rocc = occL[wid][r] != 0u;

#pragma unroll
  for (int nt = 0; nt < 2; ++nt) {
    v8f c = {};
    for (int kc = 0; kc < 16; ++kc) {
      v16h a;
#pragma unroll
      for (int e = 0; e < 16; ++e) {
        int k = kc * 32 + kmap(e, half_);
        a[e] = rocc ? (_Float16)mono2f(maxenc[rowBase + k]) : (_Float16)0.f;
      }
      v16h bw = *(const v16h*)(wfo + ((size_t)((kc * 2 + nt) * 32 + lane) << 4));
      c = __builtin_amdgcn_wmma_f32_16x16x32_f16(false, a, false, bw, (short)0, c, false, false);
    }
    int ch = nt * 16 + col;
    float bb = bzo[ch];
#pragma unroll
    for (int v = 0; v < 8; ++v) {
      int vv = v0 + mbase + v;
      float y = c[v] + bb;
      y = (y >= 0.f) ? y : SLOPE * y;
      if (!occL[wid][mbase + v]) y = 0.f;
      int b = vv / (HH * WWID);
      int rem = vv - b * (HH * WWID);
      int hh = rem / WWID, ww = rem - hh * WWID;
      out[(((size_t)b * 32 + ch) * HH + hh) * WWID + ww] = y;  // [B,32,H,W]
    }
  }
}

// ------------------------------------------------------------------ host ----
static inline size_t alignup(size_t x) { return (x + 255) & ~(size_t)255; }

extern "C" void kernel_launch(void* const* d_in, const int* in_sizes, int n_in,
                              void* d_out, int out_size, void* d_ws, size_t ws_size,
                              hipStream_t stream) {
  (void)n_in; (void)out_size; (void)ws_size;
  const float* pt_fea = (const float*)d_in[0];
  const int*   xy     = (const int*)d_in[1];
  const int*   bidx   = (const int*)d_in[2];
  int N = in_sizes[0] / 9;

  const float* bn[4][4];
  for (int i = 0; i < 4; ++i)
    for (int j = 0; j < 4; ++j) bn[i][j] = (const float*)d_in[3 + i * 4 + j];
  const float* Wl[4]; const float* bl[4];
  for (int i = 0; i < 4; ++i) { Wl[i] = (const float*)d_in[19 + 2 * i]; bl[i] = (const float*)d_in[20 + 2 * i]; }
  const float* Wo = (const float*)d_in[27];
  const float* bo = (const float*)d_in[28];

  char* ws = (char*)d_ws;
  size_t off = 0;
  unsigned* cnt    = (unsigned*)(ws + off); off = alignup(off + (size_t)NSEG * 4);
  unsigned* maxenc = (unsigned*)(ws + off); off = alignup(off + (size_t)NSEG * 512 * 4);
  _Float16* wf1 = (_Float16*)(ws + off); off = alignup(off + 2048 * 2);
  _Float16* wf2 = (_Float16*)(ws + off); off = alignup(off + 8192 * 2);
  _Float16* wf3 = (_Float16*)(ws + off); off = alignup(off + 32768 * 2);
  _Float16* wf4 = (_Float16*)(ws + off); off = alignup(off + 131072 * 2);
  _Float16* wfo = (_Float16*)(ws + off); off = alignup(off + 16384 * 2);
  float* bz1 = (float*)(ws + off); off = alignup(off + 64 * 4);
  float* bz2 = (float*)(ws + off); off = alignup(off + 128 * 4);
  float* bz3 = (float*)(ws + off); off = alignup(off + 256 * 4);
  float* bz4 = (float*)(ws + off); off = alignup(off + 512 * 4);
  float* bzo = (float*)(ws + off); off = alignup(off + 32 * 4);

  init_ws<<<2048, 256, 0, stream>>>(cnt, (long long)NSEG);
  init_ws<<<8192, 256, 0, stream>>>(maxenc, (long long)NSEG * 512);

  auto launch_prep = [&](const float* Ws, const float* bs, int Kreal, int KC, int NT, int Nout,
                         const float* const* pre, const float* const* post,
                         _Float16* wdst, float* bdst) {
    int fragCount = KC * NT * 512;
    int blocks = (fragCount + 255) / 256;
    prep_layer<<<blocks, 256, 0, stream>>>(Ws, bs, Kreal, KC, NT, Nout,
        pre ? pre[0] : nullptr, pre ? pre[1] : nullptr, pre ? pre[2] : nullptr, pre ? pre[3] : nullptr,
        post ? post[0] : nullptr, post ? post[1] : nullptr, post ? post[2] : nullptr, post ? post[3] : nullptr,
        wdst, bdst);
  };
  launch_prep(Wl[0], bl[0],   9, 1,  4,  64, bn[0], bn[1], wf1, bz1);
  launch_prep(Wl[1], bl[1],  64, 2,  8, 128, nullptr, bn[2], wf2, bz2);
  launch_prep(Wl[2], bl[2], 128, 4, 16, 256, nullptr, bn[3], wf3, bz3);
  launch_prep(Wl[3], bl[3], 256, 8, 32, 512, nullptr, nullptr, wf4, bz4);
  launch_prep(Wo,    bo,    512, 16, 2,  32, nullptr, nullptr, wfo, bzo);

  int tiles = (N + 15) / 16;
  int mblocks = (tiles + MLP_WAVES - 1) / MLP_WAVES;
  mlp_scatter<<<mblocks, 128, 0, stream>>>(pt_fea, xy, bidx, N, cnt, maxenc,
                                           wf1, wf2, wf3, wf4, bz1, bz2, bz3, bz4);

  bev_gemm<<<NSEG / (16 * BEV_WAVES), 128, 0, stream>>>(cnt, maxenc, wfo, bzo, (float*)d_out);
}
